// Transformer_80393197847056
// MI455X (gfx1250) — matrix operand order repaired and checked by reading, not compiled, not run on hardware
//
#include <hip/hip_runtime.h>
#include <math.h>

// ---------------- problem constants (match reference) ----------------
#define VOCAB   32000
#define DMODEL  1024
#define NHEAD   16
#define DKH     64          // DMODEL / NHEAD
#define FFDIM   4096
#define NLE     6
#define NLD     6
#define PAD_IDX 0
#define BSZ     4
#define SEQ     512
#define ROWS    (BSZ * SEQ) // 2048

typedef long long ll;
typedef unsigned short u16_t;
typedef __attribute__((ext_vector_type(4)))  __bf16         v4bf;
typedef __attribute__((ext_vector_type(8)))  unsigned short v8us;
typedef __attribute__((ext_vector_type(16))) unsigned short v16u;
typedef __attribute__((ext_vector_type(16))) __bf16         v16bf;
typedef __attribute__((ext_vector_type(8)))  float          v8f;

// fp32 -> bf16 via native conversion (RNE); lowers to v_cvt_pk_bf16_f32-class ops
__device__ __forceinline__ ushort4 cvt4(float4 f) {
    v4bf b;
    b[0] = (__bf16)f.x; b[1] = (__bf16)f.y;
    b[2] = (__bf16)f.z; b[3] = (__bf16)f.w;
    return __builtin_bit_cast(ushort4, b);
}

// Load a 16-element bf16 fragment as two 16-byte LDS reads (ds_load_b128)
// located at p[0..7] and p[gap..gap+7].
__device__ __forceinline__ v16bf load_frag(const u16_t* p, int gap) {
    v8us lo = *(const v8us*)p;
    v8us hi = *(const v8us*)(p + gap);
    v16u u  = __builtin_shufflevector(lo, hi, 0, 1, 2, 3, 4, 5, 6, 7,
                                              8, 9, 10, 11, 12, 13, 14, 15);
    return __builtin_bit_cast(v16bf, u);
}

#define WMMA_BF16(a, b, c) \
    __builtin_amdgcn_wmma_f32_16x16x32_bf16(false, (a), false, (b), (short)0, (c), false, false)

// ---------------- WMMA tiled GEMM: C = act(alpha * A@B(^T) + bias) ----------------
// BM=128, BN=64, BK=64; 256 threads = 8 wave32 waves in a 4(m) x 2(n) grid;
// each wave owns a 32x32 tile -> 4 f32 accumulators, 8 WMMAs per staged K-tile.
// Double-buffered LDS with register staging: one __syncthreads per K-tile,
// global b128 loads for tile k+1 overlap the 8 WMMAs of tile k.
// Strided-batched via grid.z -> (zb = z/Hdim, zh = z%Hdim).
// Requirements: K % 64 == 0 (true for 64/512/1024/4096); rows 16B-aligned.
#define BM 128
#define BN 64
#define BK 64
#define GT 256
#define LDT 72   // padded row stride (ushorts): 144 B, multiple of 16 B

__global__ __launch_bounds__(GT)
void wmma_gemm(const float* __restrict__ Ab, const float* __restrict__ Bb,
               const float* __restrict__ bias, float* __restrict__ Cb,
               int M, int N, int K,
               ll lda, ll ldb, ll ldc,
               int transB, float alpha, int relu, int Hdim,
               ll sAb, ll sAh, ll sBb, ll sBh, ll sCb, ll sCh)
{
    __shared__ u16_t As [2][BM][LDT];   // A tiles, [buf][row][k]   (36 KB)
    __shared__ u16_t Bst[2][BN][LDT];   // B tiles transposed       (18 KB)

    const int tid  = threadIdx.x;
    const int lane = tid & 31;
    const int wave = tid >> 5;
    const int wm   = wave & 3;       // 4 row groups of 32
    const int wn   = wave >> 2;      // 2 col groups of 32
    const int hf   = lane >> 4;      // lane half (ISA fragment layout)
    const int l16  = lane & 15;

    const int zb = (int)blockIdx.z / Hdim;
    const int zh = (int)blockIdx.z % Hdim;
    const float* A = Ab + (ll)zb * sAb + (ll)zh * sAh;
    const float* B = Bb + (ll)zb * sBb + (ll)zh * sBh;
    float*       C = Cb + (ll)zb * sCb + (ll)zh * sCh;

    const int rowBase = blockIdx.y * BM;
    const int colBase = blockIdx.x * BN;

    v8f acc00 = {}, acc01 = {}, acc10 = {}, acc11 = {};

    float4 ra[8];   // staged A: BM*BK/4/GT = 8 float4 per thread
    float4 rb[4];   // staged B: BN*BK/4/GT = 4 float4 per thread

    // ---- stage global -> registers for K-tile at k0 ----
    auto loadRegs = [&](int k0) {
        #pragma unroll
        for (int j = 0; j < 8; ++j) {
            int g  = tid + j * GT;
            int r  = g >> 4;                 // BK/4 = 16 groups per row
            int c4 = (g & 15) << 2;
            int gr = rowBase + r;
            float4 f = make_float4(0.f, 0.f, 0.f, 0.f);
            if (gr < M) f = *(const float4*)&A[(ll)gr * lda + (k0 + c4)];
            ra[j] = f;
        }
        #pragma unroll
        for (int j = 0; j < 4; ++j) {
            int g   = tid + j * GT;
            int n   = g & (BN - 1);
            int kk4 = (g >> 6) << 2;         // (g / BN) * 4
            int gn  = colBase + n;
            int gk  = k0 + kk4;
            float4 f = make_float4(0.f, 0.f, 0.f, 0.f);
            if (gn < N) {
                if (transB) {
                    f = *(const float4*)&B[(ll)gn * ldb + gk];
                } else {
                    f.x = B[(ll)(gk    ) * ldb + gn];
                    f.y = B[(ll)(gk + 1) * ldb + gn];
                    f.z = B[(ll)(gk + 2) * ldb + gn];
                    f.w = B[(ll)(gk + 3) * ldb + gn];
                }
            }
            rb[j] = f;
        }
    };

    // ---- convert + spill registers -> LDS buffer ----
    auto storeLDS = [&](int buf) {
        #pragma unroll
        for (int j = 0; j < 8; ++j) {
            int g  = tid + j * GT;
            int r  = g >> 4;
            int c4 = (g & 15) << 2;
            *(ushort4*)&As[buf][r][c4] = cvt4(ra[j]);
        }
        #pragma unroll
        for (int j = 0; j < 4; ++j) {
            int g   = tid + j * GT;
            int n   = g & (BN - 1);
            int kk4 = (g >> 6) << 2;
            *(ushort4*)&Bst[buf][n][kk4] = cvt4(rb[j]);
        }
    };

    const int nk = K / BK;                 // K % 64 == 0 by construction
    loadRegs(0);
    storeLDS(0);

    for (int kt = 0; kt < nk; ++kt) {
        __syncthreads();                   // buffer `cur` visible to all waves
        const int cur = kt & 1;
        if (kt + 1 < nk) loadRegs((kt + 1) * BK);   // overlap HBM with WMMA
        if (kt + 2 < nk) {                           // prefetch tile k+2
            int pr = rowBase + (tid & (BM - 1));
            if (pr < M) __builtin_prefetch(&A[(ll)pr * lda + (kt + 2) * BK], 0, 1);
        }
        // ---- two K-steps of 32: 16x ds_load_b128, 8 WMMAs ----
        #pragma unroll
        for (int ks = 0; ks < BK; ks += 32) {
            const u16_t* a0p = &As [cur][wm * 32 + l16     ][ks + hf * 8];
            const u16_t* a1p = &As [cur][wm * 32 + 16 + l16][ks + hf * 8];
            const u16_t* b0p = &Bst[cur][wn * 32 + l16     ][ks + hf * 8];
            const u16_t* b1p = &Bst[cur][wn * 32 + 16 + l16][ks + hf * 8];
            v16bf a0 = load_frag(a0p, 16);
            v16bf a1 = load_frag(a1p, 16);
            v16bf b0 = load_frag(b0p, 16);
            v16bf b1 = load_frag(b1p, 16);
            acc00 = WMMA_BF16(a0, b0, acc00);
            acc01 = WMMA_BF16(a0, b1, acc01);
            acc10 = WMMA_BF16(a1, b0, acc10);
            acc11 = WMMA_BF16(a1, b1, acc11);
        }
        if (kt + 1 < nk) storeLDS(cur ^ 1);          // fill the other buffer
    }

    // ---- epilogue: C/D layout VGPR r -> M = r (lanes 0-15) / r+8 (lanes 16-31) ----
    #pragma unroll
    for (int sn = 0; sn < 2; ++sn) {
        int col = colBase + wn * 32 + sn * 16 + l16;
        if (col >= N) continue;
        float bv = bias ? bias[col] : 0.f;
        const v8f* accs[2] = { sn ? &acc01 : &acc00, sn ? &acc11 : &acc10 };
        #pragma unroll
        for (int sm = 0; sm < 2; ++sm) {
            #pragma unroll
            for (int r = 0; r < 8; ++r) {
                int row = rowBase + wm * 32 + sm * 16 + hf * 8 + r;
                if (row >= M) continue;
                float v = (*accs[sm])[r] * alpha + bv;
                if (relu) v = fmaxf(v, 0.f);
                C[(ll)row * ldc + col] = v;
            }
        }
    }
}

// ---------------- masked softmax over scores [B,H,Sq,Sk] ----------------
__global__ __launch_bounds__(256)
void masked_softmax(float* __restrict__ scores, const int* __restrict__ keytok,
                    int Sq, int Sk, int causal)
{
    const int q = blockIdx.x, h = blockIdx.y, b = blockIdx.z;
    float* row = scores + (((ll)b * gridDim.y + h) * Sq + q) * (ll)Sk;
    const int* kt = keytok + (ll)b * Sk;
    __shared__ float red[256];
    const int tid = threadIdx.x;

    float mx = -3.4e38f;
    for (int k = tid; k < Sk; k += 256) {
        bool keep = (kt[k] != PAD_IDX) && (!causal || k <= q);
        float s = keep ? row[k] : -3.4e38f;
        row[k] = s;
        mx = fmaxf(mx, s);
    }
    red[tid] = mx; __syncthreads();
    for (int st = 128; st > 0; st >>= 1) {
        if (tid < st) red[tid] = fmaxf(red[tid], red[tid + st]);
        __syncthreads();
    }
    mx = red[0]; __syncthreads();

    float sum = 0.f;
    for (int k = tid; k < Sk; k += 256) {
        float e = __expf(row[k] - mx);
        row[k] = e;
        sum += e;
    }
    red[tid] = sum; __syncthreads();
    for (int st = 128; st > 0; st >>= 1) {
        if (tid < st) red[tid] += red[tid + st];
        __syncthreads();
    }
    float inv = 1.f / red[0];
    for (int k = tid; k < Sk; k += 256) row[k] *= inv;
}

// ---------------- fused residual-add + LayerNorm (in place on x) ----------------
__global__ __launch_bounds__(256)
void add_layernorm(float* __restrict__ x, const float* __restrict__ r,
                   const float* __restrict__ g, const float* __restrict__ bta)
{
    const ll row = blockIdx.x;
    float* xp = x + row * DMODEL;
    const float* rp = r + row * DMODEL;
    const int tid = threadIdx.x;
    __shared__ float red[256];

    float vals[4];
    float s = 0.f;
    #pragma unroll
    for (int i = 0; i < 4; ++i) {
        int d = tid + i * 256;
        vals[i] = xp[d] + rp[d];
        s += vals[i];
    }
    red[tid] = s; __syncthreads();
    for (int st = 128; st > 0; st >>= 1) {
        if (tid < st) red[tid] += red[tid + st];
        __syncthreads();
    }
    float mean = red[0] * (1.f / DMODEL); __syncthreads();

    float s2 = 0.f;
    #pragma unroll
    for (int i = 0; i < 4; ++i) { float d = vals[i] - mean; s2 += d * d; }
    red[tid] = s2; __syncthreads();
    for (int st = 128; st > 0; st >>= 1) {
        if (tid < st) red[tid] += red[tid + st];
        __syncthreads();
    }
    float rstd = rsqrtf(red[0] * (1.f / DMODEL) + 1e-5f);
    #pragma unroll
    for (int i = 0; i < 4; ++i) {
        int d = tid + i * 256;
        xp[d] = (vals[i] - mean) * rstd * g[d] + bta[d];
    }
}

// ---------------- embedding gather * sqrt(D) + sinusoidal posenc ----------------
__global__ __launch_bounds__(256)
void embed_pe(const int* __restrict__ tok, const float* __restrict__ emb,
              float* __restrict__ out)
{
    const int row = blockIdx.x;          // b*SEQ + s
    const int s   = row % SEQ;
    const int t   = tok[row];
    const float pos = (float)s;
    const float nl  = -9.2103403720f / (float)DMODEL;  // -ln(10000)/D
    for (int i = threadIdx.x; i < DMODEL; i += 256) {
        float div = __expf((float)(i & ~1) * nl);
        float pe  = (i & 1) ? __cosf(pos * div) : __sinf(pos * div);
        out[(ll)row * DMODEL + i] = emb[(ll)t * DMODEL + i] * 32.0f + pe; // sqrt(1024)=32
    }
}

// ---------------- host-side launch helper ----------------
static void gemm(hipStream_t st, const float* A, const float* B, const float* bias,
                 float* C, int M, int N, int K, ll lda, ll ldb, ll ldc,
                 int transB, float alpha, int relu,
                 int batches, int Hdim,
                 ll sAb, ll sAh, ll sBb, ll sBh, ll sCb, ll sCh)
{
    dim3 grid((N + BN - 1) / BN, (M + BM - 1) / BM, batches);
    wmma_gemm<<<grid, GT, 0, st>>>(A, B, bias, C, M, N, K, lda, ldb, ldc,
                                   transB, alpha, relu, Hdim,
                                   sAb, sAh, sBb, sBh, sCb, sCh);
}

extern "C" void kernel_launch(void* const* d_in, const int* in_sizes, int n_in,
                              void* d_out, int out_size, void* d_ws, size_t ws_size,
                              hipStream_t stream)
{
    const int*   src        = (const int*)  d_in[0];
    const int*   tgt        = (const int*)  d_in[1];
    const float* emb        = (const float*)d_in[2];
    const float* enc_attn_w = (const float*)d_in[3];
    const float* enc_attn_b = (const float*)d_in[4];
    const float* enc_ff_w1  = (const float*)d_in[5];
    const float* enc_ff_b1  = (const float*)d_in[6];
    const float* enc_ff_w2  = (const float*)d_in[7];
    const float* enc_ff_b2  = (const float*)d_in[8];
    const float* enc_ln_g   = (const float*)d_in[9];
    const float* enc_ln_b   = (const float*)d_in[10];
    const float* dec_attn_w = (const float*)d_in[11];
    const float* dec_attn_b = (const float*)d_in[12];
    const float* dec_ff_w1  = (const float*)d_in[13];
    const float* dec_ff_b1  = (const float*)d_in[14];
    const float* dec_ff_w2  = (const float*)d_in[15];
    const float* dec_ff_b2  = (const float*)d_in[16];
    const float* dec_ln_g   = (const float*)d_in[17];
    const float* dec_ln_b   = (const float*)d_in[18];
    const float* out_w      = (const float*)d_in[19];
    const float* out_b      = (const float*)d_in[20];
    float* out = (float*)d_out;

    // workspace carve-up (fp32), ~160 MB total
    float* ws = (float*)d_ws;
    size_t o = 0;
    float* x   = ws + o; o += (size_t)ROWS * DMODEL;   // encoder act / memory
    float* y   = ws + o; o += (size_t)ROWS * DMODEL;   // decoder act
    float* q   = ws + o; o += (size_t)ROWS * DMODEL;
    float* k   = ws + o; o += (size_t)ROWS * DMODEL;
    float* v   = ws + o; o += (size_t)ROWS * DMODEL;
    float* ctx = ws + o; o += (size_t)ROWS * DMODEL;
    float* tmp = ws + o; o += (size_t)ROWS * DMODEL;
    float* ffh = ws + o; o += (size_t)ROWS * FFDIM;
    float* sc  = ws + o; o += (size_t)BSZ * NHEAD * SEQ * SEQ;

    const ll DD  = (ll)DMODEL * DMODEL;
    const ll sQb = (ll)SEQ * DMODEL;          // per-b stride in q/k/v/ctx
    const ll sSb = (ll)NHEAD * SEQ * SEQ;     // per-b stride in scores
    const ll sSh = (ll)SEQ * SEQ;             // per-h stride in scores
    const float iskd = 0.125f;                // 1/sqrt(64)

    // ---------------- encoder ----------------
    embed_pe<<<ROWS, 256, 0, stream>>>(src, emb, x);
    for (int i = 0; i < NLE; ++i) {
        const float* W  = enc_attn_w + (size_t)i * 4 * DD;
        const float* Wb = enc_attn_b + (size_t)i * 4 * DMODEL;
        gemm(stream, x, W,          Wb,            q, ROWS, DMODEL, DMODEL, DMODEL, DMODEL, DMODEL, 0, 1.f, 0, 1, 1, 0,0,0,0,0,0);
        gemm(stream, x, W + DD,     Wb + DMODEL,   k, ROWS, DMODEL, DMODEL, DMODEL, DMODEL, DMODEL, 0, 1.f, 0, 1, 1, 0,0,0,0,0,0);
        gemm(stream, x, W + 2 * DD, Wb + 2*DMODEL, v, ROWS, DMODEL, DMODEL, DMODEL, DMODEL, DMODEL, 0, 1.f, 0, 1, 1, 0,0,0,0,0,0);
        // scores[b,h] = Q @ K^T / sqrt(dk)
        gemm(stream, q, k, nullptr, sc, SEQ, SEQ, DKH, DMODEL, DMODEL, SEQ, 1, iskd, 0,
             BSZ * NHEAD, NHEAD, sQb, DKH, sQb, DKH, sSb, sSh);
        masked_softmax<<<dim3(SEQ, NHEAD, BSZ), 256, 0, stream>>>(sc, src, SEQ, SEQ, 0);
        // ctx[b,h] = attn @ V
        gemm(stream, sc, v, nullptr, ctx, SEQ, DKH, SEQ, SEQ, DMODEL, DMODEL, 0, 1.f, 0,
             BSZ * NHEAD, NHEAD, sSb, sSh, sQb, DKH, sQb, DKH);
        gemm(stream, ctx, W + 3 * DD, Wb + 3 * DMODEL, tmp, ROWS, DMODEL, DMODEL, DMODEL, DMODEL, DMODEL, 0, 1.f, 0, 1, 1, 0,0,0,0,0,0);
        add_layernorm<<<ROWS, 256, 0, stream>>>(x, tmp,
            enc_ln_g + (size_t)i * 2 * DMODEL, enc_ln_b + (size_t)i * 2 * DMODEL);
        // FFN
        gemm(stream, x, enc_ff_w1 + (size_t)i * DMODEL * FFDIM, enc_ff_b1 + (size_t)i * FFDIM,
             ffh, ROWS, FFDIM, DMODEL, DMODEL, FFDIM, FFDIM, 0, 1.f, 1, 1, 1, 0,0,0,0,0,0);
        gemm(stream, ffh, enc_ff_w2 + (size_t)i * FFDIM * DMODEL, enc_ff_b2 + (size_t)i * DMODEL,
             tmp, ROWS, DMODEL, FFDIM, FFDIM, DMODEL, DMODEL, 0, 1.f, 0, 1, 1, 0,0,0,0,0,0);
        add_layernorm<<<ROWS, 256, 0, stream>>>(x, tmp,
            enc_ln_g + (size_t)i * 2 * DMODEL + DMODEL, enc_ln_b + (size_t)i * 2 * DMODEL + DMODEL);
    }

    // ---------------- decoder ----------------
    embed_pe<<<ROWS, 256, 0, stream>>>(tgt, emb, y);
    for (int i = 0; i < NLD; ++i) {
        const float* W  = dec_attn_w + (size_t)i * 8 * DD;
        const float* Wb = dec_attn_b + (size_t)i * 8 * DMODEL;
        // self-attention (causal, key padding from tgt)
        gemm(stream, y, W,          Wb,            q, ROWS, DMODEL, DMODEL, DMODEL, DMODEL, DMODEL, 0, 1.f, 0, 1, 1, 0,0,0,0,0,0);
        gemm(stream, y, W + DD,     Wb + DMODEL,   k, ROWS, DMODEL, DMODEL, DMODEL, DMODEL, DMODEL, 0, 1.f, 0, 1, 1, 0,0,0,0,0,0);
        gemm(stream, y, W + 2 * DD, Wb + 2*DMODEL, v, ROWS, DMODEL, DMODEL, DMODEL, DMODEL, DMODEL, 0, 1.f, 0, 1, 1, 0,0,0,0,0,0);
        gemm(stream, q, k, nullptr, sc, SEQ, SEQ, DKH, DMODEL, DMODEL, SEQ, 1, iskd, 0,
             BSZ * NHEAD, NHEAD, sQb, DKH, sQb, DKH, sSb, sSh);
        masked_softmax<<<dim3(SEQ, NHEAD, BSZ), 256, 0, stream>>>(sc, tgt, SEQ, SEQ, 1);
        gemm(stream, sc, v, nullptr, ctx, SEQ, DKH, SEQ, SEQ, DMODEL, DMODEL, 0, 1.f, 0,
             BSZ * NHEAD, NHEAD, sSb, sSh, sQb, DKH, sQb, DKH);
        gemm(stream, ctx, W + 3 * DD, Wb + 3 * DMODEL, tmp, ROWS, DMODEL, DMODEL, DMODEL, DMODEL, DMODEL, 0, 1.f, 0, 1, 1, 0,0,0,0,0,0);
        add_layernorm<<<ROWS, 256, 0, stream>>>(y, tmp,
            dec_ln_g + (size_t)i * 3 * DMODEL, dec_ln_b + (size_t)i * 3 * DMODEL);
        // cross-attention (keys/values from encoder memory x, key padding from src)
        gemm(stream, y, W + 4 * DD, Wb + 4 * DMODEL, q, ROWS, DMODEL, DMODEL, DMODEL, DMODEL, DMODEL, 0, 1.f, 0, 1, 1, 0,0,0,0,0,0);
        gemm(stream, x, W + 5 * DD, Wb + 5 * DMODEL, k, ROWS, DMODEL, DMODEL, DMODEL, DMODEL, DMODEL, 0, 1.f, 0, 1, 1, 0,0,0,0,0,0);
        gemm(stream, x, W + 6 * DD, Wb + 6 * DMODEL, v, ROWS, DMODEL, DMODEL, DMODEL, DMODEL, DMODEL, 0, 1.f, 0, 1, 1, 0,0,0,0,0,0);
        gemm(stream, q, k, nullptr, sc, SEQ, SEQ, DKH, DMODEL, DMODEL, SEQ, 1, iskd, 0,
             BSZ * NHEAD, NHEAD, sQb, DKH, sQb, DKH, sSb, sSh);
        masked_softmax<<<dim3(SEQ, NHEAD, BSZ), 256, 0, stream>>>(sc, src, SEQ, SEQ, 0);
        gemm(stream, sc, v, nullptr, ctx, SEQ, DKH, SEQ, SEQ, DMODEL, DMODEL, 0, 1.f, 0,
             BSZ * NHEAD, NHEAD, sSb, sSh, sQb, DKH, sQb, DKH);
        gemm(stream, ctx, W + 7 * DD, Wb + 7 * DMODEL, tmp, ROWS, DMODEL, DMODEL, DMODEL, DMODEL, DMODEL, 0, 1.f, 0, 1, 1, 0,0,0,0,0,0);
        add_layernorm<<<ROWS, 256, 0, stream>>>(y, tmp,
            dec_ln_g + (size_t)i * 3 * DMODEL + DMODEL, dec_ln_b + (size_t)i * 3 * DMODEL + DMODEL);
        // FFN
        gemm(stream, y, dec_ff_w1 + (size_t)i * DMODEL * FFDIM, dec_ff_b1 + (size_t)i * FFDIM,
             ffh, ROWS, FFDIM, DMODEL, DMODEL, FFDIM, FFDIM, 0, 1.f, 1, 1, 1, 0,0,0,0,0,0);
        gemm(stream, ffh, dec_ff_w2 + (size_t)i * FFDIM * DMODEL, dec_ff_b2 + (size_t)i * DMODEL,
             tmp, ROWS, DMODEL, FFDIM, FFDIM, DMODEL, DMODEL, 0, 1.f, 0, 1, 1, 0,0,0,0,0,0);
        add_layernorm<<<ROWS, 256, 0, stream>>>(y, tmp,
            dec_ln_g + (size_t)i * 3 * DMODEL + 2 * DMODEL, dec_ln_b + (size_t)i * 3 * DMODEL + 2 * DMODEL);
    }

    // ---------------- logits: [2048,1024] @ [1024,32000] + bias ----------------
    gemm(stream, y, out_w, out_b, out, ROWS, VOCAB, DMODEL, DMODEL, VOCAB, VOCAB, 0, 1.f, 0, 1, 1, 0,0,0,0,0,0);
}
